// SparseDecoder_35218731827741
// MI455X (gfx1250) — matrix operand order repaired and checked by reading, not compiled, not run on hardware
//
#include <hip/hip_runtime.h>

// ---------------------------------------------------------------------------
// CDNA5 (gfx1250, wave32) implementation of the 12-layer linear-attention
// transformer. All matmuls run on v_wmma_f32_16x16x32_f16 (f16 in, f32 acc).
// Weights are pre-swizzled into WMMA B-fragment order (one contiguous 32B
// load per lane); activations are kept in row-major f16 with K padded to a
// multiple of 32 so A-fragments are two contiguous 16B loads per lane.
// GEMM waves compute a 16x64 strip; fragment loads are hoisted ahead of the
// WMMA group so each s_wait_loadcnt covers a burst of independent WMMAs.
// Attention streams over key tiles (no NxN score materialization).
// ---------------------------------------------------------------------------

typedef _Float16 h8   __attribute__((ext_vector_type(8)));
typedef _Float16 v16h __attribute__((ext_vector_type(16)));
typedef float    v8f  __attribute__((ext_vector_type(8)));

#define BS      16
#define SEQ     1024
#define EMB     128
#define D1      63
#define DP      64
#define NH      4
#define HID     512
#define NLAYER  12
#define MROWS   (BS * SEQ)   // 16384
#define LN_EPS  1e-5f

// ---------------------------------------------------------------------------
// WMMA helpers
// ---------------------------------------------------------------------------
__device__ __forceinline__ v8f wmma_f16(v16h a, v16h b, v8f c) {
  // D = A(16x32 f16) * B(32x16 f16) + C(16x16 f32)
  return __builtin_amdgcn_wmma_f32_16x16x32_f16(false, a, false, b, (short)0, c,
                                                false, false);
}

// A-fragment (16x32 f16): lane L: row = L&15; element e -> K = (L>>4)*16+e
// : two contiguous 8-half (16B) loads.
__device__ __forceinline__ v16h load_afrag(const _Float16* __restrict__ base,
                                           int rs, int lane) {
  int r = lane & 15, hi = lane >> 4;
  const _Float16* p = base + (size_t)r * rs + hi * 16;
  h8 lo = *(const h8*)(p);
  h8 up = *(const h8*)(p + 8);
  v16h a;
#pragma unroll
  for (int i = 0; i < 8; ++i) { a[i] = lo[i]; a[i + 8] = up[i]; }
  return a;
}

// B-fragment (32x16) where B[k][n] = mat[n][k] (k contiguous along a row of mat).
// lane L: n = L&15, k = e + (L>>4)*16 : one contiguous 16-half run per lane.
__device__ __forceinline__ v16h load_bfrag_rows(const _Float16* __restrict__ base,
                                                int rs, int lane) {
  int n = lane & 15, hi = lane >> 4;
  const _Float16* p = base + (size_t)n * rs + hi * 16;
  h8 lo = *(const h8*)(p);
  h8 up = *(const h8*)(p + 8);
  v16h b;
#pragma unroll
  for (int i = 0; i < 8; ++i) { b[i] = lo[i]; b[i + 8] = up[i]; }
  return b;
}

// ---------------------------------------------------------------------------
// Generic GEMM: C[M,Nn](f32 and/or f16) = A[M,K](f16, row major) * Bsw + bias
// Bsw is fragment-swizzled: flat index ((kt*(Nn/16)+nt)*32 + lane)*16 + e.
// One wave per 16x64 output strip (4 accumulators, A reused across 4 WMMAs);
// 4 waves per block. Requires Nn % 64 == 0, K % 32 == 0.
// ---------------------------------------------------------------------------
__global__ __launch_bounds__(128)
void gemm_kernel(const _Float16* __restrict__ A, int lda,
                 const _Float16* __restrict__ Bsw,
                 float* __restrict__ Cf, _Float16* __restrict__ Ch, int ldc,
                 int M, int Nn, int K,
                 const float* __restrict__ bias, int do_relu) {
  int lane = threadIdx.x & 31;
  int wave = threadIdx.x >> 5;
  int ntn  = Nn >> 4;   // 16-wide B tiles per K-slab
  int ng4  = Nn >> 6;   // 64-wide output groups
  int tile = blockIdx.x * 4 + wave;
  if (tile >= (M >> 4) * ng4) return;
  int mt = tile / ng4;
  int ng = tile - mt * ng4;
  int nkt = K >> 5;

  v8f acc[4];
#pragma unroll
  for (int s = 0; s < 4; ++s)
#pragma unroll
    for (int i = 0; i < 8; ++i) acc[s][i] = 0.0f;

  const _Float16* Ab = A + (size_t)mt * 16 * lda;
  const _Float16* Bb = Bsw + ((size_t)ng * 4 * 32 + lane) * 16;
  for (int kt = 0; kt < nkt; ++kt) {
    // hoist all fragment loads: one clause, one wait, 4 independent WMMAs
    v16h a = load_afrag(Ab + kt * 32, lda, lane);
    const _Float16* bp = Bb + (size_t)kt * ntn * 32 * 16;
    v16h bfr[4];
#pragma unroll
    for (int s = 0; s < 4; ++s)
      bfr[s] = *(const v16h*)(bp + (size_t)s * 32 * 16);
#pragma unroll
    for (int s = 0; s < 4; ++s)
      acc[s] = wmma_f16(a, bfr[s], acc[s]);
  }

  int r = lane & 15, hi = lane >> 4;
#pragma unroll
  for (int s = 0; s < 4; ++s) {
    int col = ng * 64 + s * 16 + r;
    float bv = bias ? bias[col] : 0.0f;
#pragma unroll
    for (int e = 0; e < 8; ++e) {
      int row = mt * 16 + hi * 8 + e;
      float v = acc[s][e] + bv;
      if (do_relu) v = v > 0.0f ? v : 0.0f;
      if (Cf) Cf[(size_t)row * ldc + col] = v;
      if (Ch) Ch[(size_t)row * ldc + col] = (_Float16)v;
    }
  }
}

// ---------------------------------------------------------------------------
// Fused causal linear attention for one (batch, head, 16-query tile).
//   S = relu(Q K^T) * hm^2 / (i+1)  (for j<=i, else 0);  O += S * V
// Q: [.., rows, q_rs] f16 (head at q_hoff). K likewise. Vt: [b][h][DH][SEQ].
// O: [b][h][SEQ][128] f32 (first DH cols valid).
// One wave per block; S re-shuffled C-layout -> A-layout through 1KB of LDS.
// ---------------------------------------------------------------------------
template <int DH>
__global__ __launch_bounds__(32)
void attn_kernel(const _Float16* __restrict__ Q, int q_bs, int q_hoff, int q_rs,
                 const _Float16* __restrict__ Km, int k_bs, int k_hoff, int k_rs,
                 const _Float16* __restrict__ Vt,
                 float* __restrict__ O,
                 const float* __restrict__ hm) {
  constexpr int NF  = DH / 32;   // QK^T k-steps
  constexpr int NVT = DH / 16;   // output feature tiles
  int lane = threadIdx.x;
  int qt = blockIdx.x;
  int b = blockIdx.y >> 2, h = blockIdx.y & 3;

  const _Float16* qp = Q + (size_t)b * q_bs + h * q_hoff + (size_t)qt * 16 * q_rs;
  const _Float16* kp = Km + (size_t)b * k_bs + h * k_hoff;
  const _Float16* vp = Vt + ((size_t)(b * NH + h)) * DH * SEQ;
  float m = hm[h];
  float scale2 = m * m;

  v16h qa[NF];
#pragma unroll
  for (int f = 0; f < NF; ++f) qa[f] = load_afrag(qp + f * 32, q_rs, lane);

  v8f oc[NVT];
#pragma unroll
  for (int t = 0; t < NVT; ++t)
#pragma unroll
    for (int i = 0; i < 8; ++i) oc[t][i] = 0.0f;

  __shared__ _Float16 Sl[16][32];
  int r = lane & 15, hi = lane >> 4;

  for (int kt0 = 0; kt0 <= qt; kt0 += 2) {
#pragma unroll
    for (int t = 0; t < 2; ++t) {
      int kt = kt0 + t;
      v8f s;
#pragma unroll
      for (int i = 0; i < 8; ++i) s[i] = 0.0f;
      const _Float16* kb = kp + (size_t)kt * 16 * k_rs;
      // hoist K fragments, then burst the WMMAs
      v16h kfr[NF];
#pragma unroll
      for (int f = 0; f < NF; ++f)
        kfr[f] = load_bfrag_rows(kb + f * 32, k_rs, lane);
#pragma unroll
      for (int f = 0; f < NF; ++f)
        s = wmma_f16(qa[f], kfr[f], s);
      // relu + triangular count-normalization, store to LDS as f16
      int j = kt * 16 + r;  // key index (C-layout col = lane&15)
#pragma unroll
      for (int e = 0; e < 8; ++e) {
        int i = qt * 16 + hi * 8 + e;  // query index (C-layout row)
        float v = s[e];
        v = v > 0.0f ? v * scale2 / (float)(i + 1) : 0.0f;
        if (j > i) v = 0.0f;
        Sl[hi * 8 + e][t * 16 + r] = (_Float16)v;
      }
    }
    __syncthreads();
    // S A-fragment over 32 keys from LDS (contiguous 8-half runs)
    v16h sa;
#pragma unroll
    for (int i = 0; i < 8; ++i) {
      sa[i]     = Sl[r][hi * 16 + i];
      sa[i + 8] = Sl[r][hi * 16 + 8 + i];
    }
    __syncthreads();
    // O += S * V  (V^T stored so B-fragment loads are contiguous).
    // V fragments hoisted in groups of 4 -> one wait per 4 WMMAs.
    const _Float16* vb0 = vp + kt0 * 16 + hi * 16;
#pragma unroll
    for (int t0 = 0; t0 < NVT; t0 += 4) {
      v16h vfr[4];
#pragma unroll
      for (int u = 0; u < 4; ++u) {
        const _Float16* p = vb0 + (size_t)((t0 + u) * 16 + r) * SEQ;
        h8 lo = *(const h8*)(p);
        h8 up = *(const h8*)(p + 8);
#pragma unroll
        for (int i = 0; i < 8; ++i) { vfr[u][i] = lo[i]; vfr[u][i + 8] = up[i]; }
      }
#pragma unroll
      for (int u = 0; u < 4; ++u)
        oc[t0 + u] = wmma_f16(sa, vfr[u], oc[t0 + u]);
    }
  }

  float* op = O + (((size_t)(b * NH + h)) * SEQ + (size_t)qt * 16) * EMB;
#pragma unroll
  for (int t = 0; t < NVT; ++t)
#pragma unroll
    for (int e = 0; e < 8; ++e)
      op[(hi * 8 + e) * EMB + t * 16 + r] = oc[t][e];
}

// ---------------------------------------------------------------------------
// Residual add + LayerNorm (wave per row) + emit f16 views for next layer.
// mode 0: H += src[row,128]              (MLP residual)
// mode 1: H[64..126] += sum_h O[..,0..62] (non-last attention residual)
// mode 2: H += sum_h O[..,0..127]         (last-layer attention residual)
// mode 3: no add, no LN (just produce Hf16/Hfeat/Hlab views)
// ---------------------------------------------------------------------------
__global__ __launch_bounds__(128)
void resln_kernel(float* __restrict__ H, const float* __restrict__ src,
                  const float* __restrict__ g, const float* __restrict__ bet,
                  _Float16* __restrict__ Hf, _Float16* __restrict__ Hfeat,
                  _Float16* __restrict__ Hlab, int mode) {
  int wave = threadIdx.x >> 5, lane = threadIdx.x & 31;
  int row = blockIdx.x * 4 + wave;
  if (row >= MROWS) return;
  int b = row >> 10, n = row & 1023;

  float x[4];
#pragma unroll
  for (int u = 0; u < 4; ++u) {
    int c = lane * 4 + u;
    float v = H[(size_t)row * EMB + c];
    if (mode == 0) {
      v += src[(size_t)row * EMB + c];
    } else if (mode == 2) {
#pragma unroll
      for (int h = 0; h < NH; ++h)
        v += src[(((size_t)(b * NH + h)) * SEQ + n) * EMB + c];
    } else if (mode == 1) {
      if (c >= DP && c < EMB - 1) {
#pragma unroll
        for (int h = 0; h < NH; ++h)
          v += src[(((size_t)(b * NH + h)) * SEQ + n) * EMB + (c - DP)];
      }
    }
    x[u] = v;
  }

  float y[4];
  if (mode != 3) {
    float s  = x[0] + x[1] + x[2] + x[3];
    float sq = x[0] * x[0] + x[1] * x[1] + x[2] * x[2] + x[3] * x[3];
#pragma unroll
    for (int mm = 16; mm >= 1; mm >>= 1) {
      s  += __shfl_xor(s,  mm, 32);
      sq += __shfl_xor(sq, mm, 32);
    }
    float mu  = s * (1.0f / EMB);
    float var = sq * (1.0f / EMB) - mu * mu;
    float inv = rsqrtf(var + LN_EPS);
#pragma unroll
    for (int u = 0; u < 4; ++u) {
      int c = lane * 4 + u;
      y[u] = (x[u] - mu) * inv * g[c] + bet[c];
    }
  } else {
#pragma unroll
    for (int u = 0; u < 4; ++u) y[u] = x[u];
  }

#pragma unroll
  for (int u = 0; u < 4; ++u) {
    int c = lane * 4 + u;
    if (mode != 3) H[(size_t)row * EMB + c] = y[u];
    _Float16 yh = (_Float16)y[u];
    Hf[(size_t)row * EMB + c] = yh;
    if (c < DP)
      Hfeat[(size_t)row * DP + c] = (c < D1) ? yh : (_Float16)0.0f;
    else
      Hlab[(size_t)row * DP + (c - DP)] = (c < EMB - 1) ? yh : (_Float16)0.0f;
  }
}

// ---------------------------------------------------------------------------
// _combine: interleave (x,0) and (x,y) rows -> zs f16 [BS][SEQ][64]
// ---------------------------------------------------------------------------
__global__ void zs_kernel(const float* __restrict__ xs, const float* __restrict__ ys,
                          _Float16* __restrict__ zs) {
  int idx = blockIdx.x * blockDim.x + threadIdx.x;
  if (idx >= BS * SEQ * DP) return;
  int d = idx & 63;
  int n = (idx >> 6) & 1023;
  int b = idx >> 16;
  int p = n >> 1, par = n & 1;
  float v;
  if (d < D1) v = xs[((size_t)b * 512 + p) * D1 + d];
  else        v = par ? ys[(size_t)b * 512 + p] : 0.0f;
  zs[idx] = (_Float16)v;
}

// ---------------------------------------------------------------------------
// f32 weights -> f16 B-fragment-swizzled (with zero padding; optional per-head
// 63->64 column re-padding for qW/vW of the non-last layers).
// ---------------------------------------------------------------------------
__global__ void convw_kernel(const float* __restrict__ W, _Float16* __restrict__ dst,
                             int Kp, int Np, int srcK, int srcN, int headmode) {
  int idx = blockIdx.x * blockDim.x + threadIdx.x;
  if (idx >= Kp * Np) return;
  int e    = idx & 15;
  int lane = (idx >> 4) & 31;
  int rest = idx >> 9;
  int ntn  = Np >> 4;
  int nt = rest % ntn;
  int kt = rest / ntn;
  int k = kt * 32 + (e + (lane >> 4) * 16);
  int c = nt * 16 + (lane & 15);
  float v = 0.0f;
  if (headmode) {
    int hh = c >> 6, d = c & 63;
    if (k < D1 && d < D1) v = W[(size_t)k * (NH * D1) + hh * D1 + d];
  } else {
    if (k < srcK && c < srcN) v = W[(size_t)k * srcN + c];
  }
  dst[idx] = (_Float16)v;
}

// V (f32 [BS*SEQ, C], head h at cols h*Dh..) -> Vt f16 [b][h][Dh][SEQ] * hm[h]
__global__ void convt_kernel(const float* __restrict__ src, _Float16* __restrict__ dst,
                             const float* __restrict__ hm, int Dh, int C) {
  int idx = blockIdx.x * blockDim.x + threadIdx.x;
  if (idx >= BS * NH * Dh * SEQ) return;
  int n = idx & 1023;
  int t = idx >> 10;
  int d = t % Dh; t /= Dh;
  int h = t & 3;
  int b = t >> 2;
  dst[idx] = (_Float16)(src[((size_t)(b * SEQ + n)) * C + h * Dh + d] * hm[h]);
}

// pred = H @ read_out_W + b, keep even rows -> out[BS][512]
__global__ void readout_kernel(const float* __restrict__ H, const float* __restrict__ w,
                               const float* __restrict__ b0, float* __restrict__ out) {
  int i = blockIdx.x * blockDim.x + threadIdx.x;
  if (i >= BS * 512) return;
  int b = i >> 9, p = i & 511;
  const float* hr = H + ((size_t)b * SEQ + 2 * p) * EMB;
  float s = b0[0];
#pragma unroll 4
  for (int c = 0; c < EMB; ++c) s += hr[c] * w[c];
  out[i] = s;
}

// ---------------------------------------------------------------------------
// Host launch
// ---------------------------------------------------------------------------
extern "C" void kernel_launch(void* const* d_in, const int* in_sizes, int n_in,
                              void* d_out, int out_size, void* d_ws, size_t ws_size,
                              hipStream_t stream) {
  (void)in_sizes; (void)n_in; (void)out_size; (void)ws_size;

  const float* xs        = (const float*)d_in[0];
  const float* ys        = (const float*)d_in[1];
  const float* head_mask = (const float*)d_in[2];
  const float* read_in_W = (const float*)d_in[3];
  const float* read_in_b = (const float*)d_in[4];
  const float* qW        = (const float*)d_in[5];
  const float* vW        = (const float*)d_in[6];
  const float* qW_last   = (const float*)d_in[7];
  const float* kW_last   = (const float*)d_in[8];
  const float* vW_last   = (const float*)d_in[9];
  const float* ln1_g     = (const float*)d_in[10];
  const float* ln1_b     = (const float*)d_in[11];
  const float* ln2_g     = (const float*)d_in[12];
  const float* ln2_b     = (const float*)d_in[13];
  const float* mlp_W1    = (const float*)d_in[14];
  const float* mlp_b1    = (const float*)d_in[15];
  const float* mlp_W2    = (const float*)d_in[16];
  const float* mlp_b2    = (const float*)d_in[17];
  const float* ro_W      = (const float*)d_in[18];
  const float* ro_b      = (const float*)d_in[19];
  float* out = (float*)d_out;

  // bump allocator over d_ws
  char* base = (char*)d_ws;
  size_t off = 0;
  auto alloc = [&](size_t bytes) -> void* {
    off = (off + 255) & ~(size_t)255;
    void* p = base + off;
    off += bytes;
    return p;
  };

  float*    H     = (float*)   alloc((size_t)MROWS * EMB * 4);
  _Float16* Hf    = (_Float16*)alloc((size_t)MROWS * EMB * 2);
  _Float16* Hfeat = (_Float16*)alloc((size_t)MROWS * DP * 2);
  _Float16* Hlab  = (_Float16*)alloc((size_t)MROWS * DP * 2);
  _Float16* zs    = (_Float16*)alloc((size_t)MROWS * DP * 2);
  _Float16* rwsw  = (_Float16*)alloc((size_t)64 * 128 * 2);
  _Float16* qsw   = (_Float16*)alloc((size_t)(NLAYER - 1) * 64 * 256 * 2);
  _Float16* vsw   = (_Float16*)alloc((size_t)(NLAYER - 1) * 64 * 256 * 2);
  _Float16* qlsw  = (_Float16*)alloc((size_t)128 * 512 * 2);
  _Float16* klsw  = (_Float16*)alloc((size_t)128 * 512 * 2);
  _Float16* vlsw  = (_Float16*)alloc((size_t)128 * 512 * 2);
  _Float16* m1sw  = (_Float16*)alloc((size_t)NLAYER * 128 * 512 * 2);
  _Float16* m2sw  = (_Float16*)alloc((size_t)NLAYER * 512 * 128 * 2);
  float*    Pbuf  = (float*)   alloc((size_t)MROWS * 512 * 4);
  _Float16* Qh    = (_Float16*)alloc((size_t)MROWS * 512 * 2);
  _Float16* Kh    = (_Float16*)alloc((size_t)MROWS * 512 * 2);
  _Float16* Vt    = (_Float16*)alloc((size_t)BS * NH * 128 * SEQ * 2);
  _Float16* Mh    = (_Float16*)alloc((size_t)MROWS * HID * 2);
  float*    O     = (float*)   alloc((size_t)BS * NH * SEQ * EMB * 4);

  auto gemm_grid = [](int M, int Nn) { return ((M >> 4) * (Nn >> 6) + 3) / 4; };
  auto elem_grid = [](long n) { return (int)((n + 255) / 256); };

  // ---- setup: zs, weight conversions ----
  zs_kernel<<<elem_grid((long)MROWS * DP), 256, 0, stream>>>(xs, ys, zs);
  convw_kernel<<<elem_grid(64L * 128), 256, 0, stream>>>(read_in_W, rwsw, 64, 128, 64, 128, 0);
  for (int l = 0; l < NLAYER - 1; ++l) {
    convw_kernel<<<elem_grid(64L * 256), 256, 0, stream>>>(
        qW + (size_t)l * D1 * NH * D1, qsw + (size_t)l * 64 * 256, 64, 256, D1, NH * D1, 1);
    convw_kernel<<<elem_grid(64L * 256), 256, 0, stream>>>(
        vW + (size_t)l * D1 * NH * D1, vsw + (size_t)l * 64 * 256, 64, 256, D1, NH * D1, 1);
  }
  convw_kernel<<<elem_grid(128L * 512), 256, 0, stream>>>(qW_last, qlsw, 128, 512, 128, 512, 0);
  convw_kernel<<<elem_grid(128L * 512), 256, 0, stream>>>(kW_last, klsw, 128, 512, 128, 512, 0);
  convw_kernel<<<elem_grid(128L * 512), 256, 0, stream>>>(vW_last, vlsw, 128, 512, 128, 512, 0);
  for (int l = 0; l < NLAYER; ++l) {
    convw_kernel<<<elem_grid(128L * 512), 256, 0, stream>>>(
        mlp_W1 + (size_t)l * 128 * 512, m1sw + (size_t)l * 128 * 512, 128, 512, 128, 512, 0);
    convw_kernel<<<elem_grid(512L * 128), 256, 0, stream>>>(
        mlp_W2 + (size_t)l * 512 * 128, m2sw + (size_t)l * 512 * 128, 512, 128, 512, 128, 0);
  }

  // ---- read_in: H = zs @ W + b, then f16 views (no LN before layer 0) ----
  gemm_kernel<<<gemm_grid(MROWS, 128), 128, 0, stream>>>(
      zs, DP, rwsw, H, nullptr, EMB, MROWS, EMB, DP, read_in_b, 0);
  resln_kernel<<<MROWS / 4, 128, 0, stream>>>(H, nullptr, nullptr, nullptr,
                                              Hf, Hfeat, Hlab, 3);

  dim3 agrid(SEQ / 16, BS * NH);

  for (int l = 0; l < NLAYER; ++l) {
    if (l < NLAYER - 1) {
      // q = Hlab @ qW (per-head dim padded 63->64), f16 direct
      gemm_kernel<<<gemm_grid(MROWS, 256), 128, 0, stream>>>(
          Hlab, DP, qsw + (size_t)l * 64 * 256, nullptr, Qh, 256, MROWS, 256, DP, nullptr, 0);
      // v = Hfeat @ vW -> f32, then transposed f16 with head_mask
      gemm_kernel<<<gemm_grid(MROWS, 256), 128, 0, stream>>>(
          Hfeat, DP, vsw + (size_t)l * 64 * 256, Pbuf, nullptr, 256, MROWS, 256, DP, nullptr, 0);
      convt_kernel<<<elem_grid((long)BS * NH * DP * SEQ), 256, 0, stream>>>(
          Pbuf, Vt, head_mask, DP, 256);
      // k = Hfeat (identity, shared across heads)
      attn_kernel<64><<<agrid, 32, 0, stream>>>(
          Qh, SEQ * 256, DP, 256,
          Hfeat, SEQ * DP, 0, DP,
          Vt, O, head_mask);
      resln_kernel<<<MROWS / 4, 128, 0, stream>>>(
          H, O, ln1_g + (size_t)l * EMB, ln1_b + (size_t)l * EMB, Hf, Hfeat, Hlab, 1);
    } else {
      gemm_kernel<<<gemm_grid(MROWS, 512), 128, 0, stream>>>(
          Hf, EMB, qlsw, nullptr, Qh, 512, MROWS, 512, EMB, nullptr, 0);
      gemm_kernel<<<gemm_grid(MROWS, 512), 128, 0, stream>>>(
          Hf, EMB, klsw, nullptr, Kh, 512, MROWS, 512, EMB, nullptr, 0);
      gemm_kernel<<<gemm_grid(MROWS, 512), 128, 0, stream>>>(
          Hf, EMB, vlsw, Pbuf, nullptr, 512, MROWS, 512, EMB, nullptr, 0);
      convt_kernel<<<elem_grid((long)BS * NH * EMB * SEQ), 256, 0, stream>>>(
          Pbuf, Vt, head_mask, EMB, 512);
      attn_kernel<128><<<agrid, 32, 0, stream>>>(
          Qh, SEQ * 512, EMB, 512,
          Kh, SEQ * 512, EMB, 512,
          Vt, O, head_mask);
      resln_kernel<<<MROWS / 4, 128, 0, stream>>>(
          H, O, ln1_g + (size_t)l * EMB, ln1_b + (size_t)l * EMB, Hf, Hfeat, Hlab, 2);
    }

    // MLP: relu(Hf @ W1 + b1) @ W2 + b2
    gemm_kernel<<<gemm_grid(MROWS, HID), 128, 0, stream>>>(
        Hf, EMB, m1sw + (size_t)l * 128 * 512, nullptr, Mh, HID,
        MROWS, HID, EMB, mlp_b1 + (size_t)l * HID, 1);
    gemm_kernel<<<gemm_grid(MROWS, EMB), 128, 0, stream>>>(
        Mh, HID, m2sw + (size_t)l * 512 * 128, Pbuf, nullptr, EMB,
        MROWS, EMB, HID, mlp_b2 + (size_t)l * EMB, 0);
    resln_kernel<<<MROWS / 4, 128, 0, stream>>>(
        H, Pbuf, ln2_g + (size_t)l * EMB, ln2_b + (size_t)l * EMB, Hf, Hfeat, Hlab, 0);
  }

  readout_kernel<<<elem_grid(BS * 512L), 256, 0, stream>>>(H, ro_W, ro_b, out);
}
